// GetTxtNet_16853451669874
// MI455X (gfx1250) — matrix operand order repaired and checked by reading, not compiled, not run on hardware
//
#include <hip/hip_runtime.h>

// ---------------------------------------------------------------------------
// GAT forward, MI455X (gfx1250, wave32). bf16 WMMA (16x16x32, fp32 accum) for
// all GEMMs; fused flash-style masked softmax-aggregation (attn never
// materialized). Workspace use ~172 MB.
// ---------------------------------------------------------------------------

#define NN   4096      // nodes
#define FIN  1386      // input features
#define FPAD 1408      // FIN padded to multiple of 32
#define HH   8         // heads
#define OO   1024      // per-head out
#define HO   8192      // H*O
#define CC   64        // final out channels

typedef __attribute__((ext_vector_type(16))) __bf16 v16bf;
typedef __attribute__((ext_vector_type(8)))  __bf16 v8bf;
typedef __attribute__((ext_vector_type(8)))  float  v8f;

__device__ __forceinline__ v16bf cat8(v8bf lo, v8bf hi) {
    return __builtin_shufflevector(lo, hi, 0,1,2,3,4,5,6,7,8,9,10,11,12,13,14,15);
}

__device__ __forceinline__ v8f wmma_bf16(v16bf a, v16bf b, v8f c) {
    return __builtin_amdgcn_wmma_f32_16x16x32_bf16(false, a, false, b, (short)0, c,
                                                   false, false);
}

__device__ __forceinline__ float lrelu(float s) { return s > 0.f ? s : 0.2f * s; }

// ---------------------------------------------------------------------------
// Conversions to bf16 (padded / transposed for K-contiguous fragment loads)
// ---------------------------------------------------------------------------
__global__ void k_convert_x(const float* __restrict__ x, __bf16* __restrict__ xb) {
    int idx = blockIdx.x * 256 + threadIdx.x;
    if (idx >= NN * FPAD) return;
    int n = idx / FPAD, f = idx % FPAD;
    xb[idx] = (__bf16)(f < FIN ? x[(size_t)n * FIN + f] : 0.f);
}

// WcatT[col][f], col = h*1024+o : B^T of layer-1 GEMM
__global__ void k_convert_W(const float* __restrict__ W, __bf16* __restrict__ WcatT) {
    int idx = blockIdx.x * 256 + threadIdx.x;
    if (idx >= HO * FPAD) return;
    int col = idx / FPAD, f = idx % FPAD;
    int h = col >> 10, o = col & 1023;
    float v = (f < FIN) ? W[((size_t)h * FIN + f) * OO + o] : 0.f;
    WcatT[idx] = (__bf16)v;
}

// WoutT[c][k] = W_out[k][c]
__global__ void k_convert_Wout(const float* __restrict__ W, __bf16* __restrict__ WoutT) {
    int idx = blockIdx.x * 256 + threadIdx.x;
    if (idx >= CC * HO) return;
    int c = idx / HO, k = idx % HO;
    WoutT[idx] = (__bf16)W[(size_t)k * CC + c];
}

// ---------------------------------------------------------------------------
// GEMM1: WhT[col][n] = sum_f xb[n][f] * WcatT[col][f]   (bf16 WMMA, f32 acc)
// block = 8 waves, each wave one 16x16 tile; block covers 16 n x 128 cols
// ---------------------------------------------------------------------------
__global__ void k_gemm1(const __bf16* __restrict__ A,   // [NN][FPAD]
                        const __bf16* __restrict__ BT,  // [HO][FPAD]
                        __bf16* __restrict__ CT) {      // [HO][NN]
    int lane = threadIdx.x & 31, w = threadIdx.x >> 5;
    int col0 = blockIdx.x * 128 + w * 16;
    int n0   = blockIdx.y * 16;
    int kbA = (lane >> 4) * 8;     // A frag: lane-half -> K 0..7 / 8..15 (+16)
    int kbB = (lane >> 4) * 8;     // B frag: lane-half -> K 0..7 / 8..15 (+16)
    const __bf16* arow = A  + (size_t)(n0 + (lane & 15)) * FPAD;
    const __bf16* brow = BT + (size_t)(col0 + (lane & 15)) * FPAD;
    v8f acc = {};
    for (int k0 = 0; k0 < FPAD; k0 += 32) {
        __builtin_prefetch(brow + k0 + 128, 0, 1);
        v16bf af = cat8(*(const v8bf*)(arow + k0 + kbA),
                        *(const v8bf*)(arow + k0 + 16 + kbA));
        v16bf bf = cat8(*(const v8bf*)(brow + k0 + kbB),
                        *(const v8bf*)(brow + k0 + 16 + kbB));
        acc = wmma_bf16(af, bf, acc);
    }
    int cl = lane & 15, rh = lane >> 4;
    for (int r = 0; r < 8; ++r) {
        int n = n0 + rh * 8 + r;
        CT[(size_t)(col0 + cl) * NN + n] = (__bf16)acc[r];
    }
}

// ---------------------------------------------------------------------------
// f1/f2 per (head, node): dot(Wh[h,n,:], a{1,2}[h,:]) — deterministic, coalesced
// grid (16, 8), block 256: wave w handles 32 consecutive n
// ---------------------------------------------------------------------------
__global__ void k_f1f2_h(const __bf16* __restrict__ WhT, const float* __restrict__ a1,
                         const float* __restrict__ a2, float* __restrict__ f1,
                         float* __restrict__ f2) {
    int h = blockIdx.y;
    int lane = threadIdx.x & 31, w = threadIdx.x >> 5;
    int n = blockIdx.x * 256 + w * 32 + lane;
    const __bf16* base = WhT + (size_t)h * OO * NN + n;
    const float* a1h = a1 + h * OO;
    const float* a2h = a2 + h * OO;
    float s1 = 0.f, s2 = 0.f;
    for (int o = 0; o < OO; ++o) {
        float v = (float)base[(size_t)o * NN];
        s1 += v * a1h[o];
        s2 += v * a2h[o];
    }
    f1[h * NN + n] = s1;
    f2[h * NN + n] = s2;
}

__global__ void k_f1f2_out(const __bf16* __restrict__ Wh2T, const float* __restrict__ a1,
                           const float* __restrict__ a2, float* __restrict__ f1,
                           float* __restrict__ f2) {
    int n = blockIdx.x * 256 + threadIdx.x;
    float s1 = 0.f, s2 = 0.f;
    for (int c = 0; c < CC; ++c) {
        float v = (float)Wh2T[(size_t)c * NN + n];
        s1 += v * a1[c];
        s2 += v * a2[c];
    }
    f1[n] = s1;
    f2[n] = s2;
}

// ---------------------------------------------------------------------------
// Masked softmax row stats: rowmax, rowsum(exp) for e = lrelu(f1[n]+f2[m])
// one block per node n; adj row staged in 16 KB LDS; wave per head
// ---------------------------------------------------------------------------
__global__ void k_rowstats(const float* __restrict__ adj, const float* __restrict__ f1,
                           const float* __restrict__ f2, float* __restrict__ rmax,
                           float* __restrict__ rsum, int Hn) {
    __shared__ float srow[NN];
    int n = blockIdx.x;
    int tid = threadIdx.x, lane = tid & 31, w = tid >> 5;
    for (int i = tid; i < NN; i += 256) srow[i] = adj[(size_t)n * NN + i];
    __syncthreads();
    for (int h = w; h < Hn; h += 8) {
        float f1v = f1[h * NN + n];
        const float* f2h = f2 + h * NN;
        float mx = -9e15f;
        for (int m = lane; m < NN; m += 32)
            if (srow[m] > 0.f) mx = fmaxf(mx, lrelu(f1v + f2h[m]));
        for (int o = 16; o > 0; o >>= 1) mx = fmaxf(mx, __shfl_xor(mx, o, 32));
        float sum = 0.f;
        for (int m = lane; m < NN; m += 32)
            if (srow[m] > 0.f) sum += __expf(lrelu(f1v + f2h[m]) - mx);
        for (int o = 16; o > 0; o >>= 1) sum += __shfl_xor(sum, o, 32);
        if (lane == 0) { rmax[h * NN + n] = mx; rsum[h * NN + n] = sum; }
    }
}

// ---------------------------------------------------------------------------
// Fused aggregation layer 1:  hcat[n][h*1024+o] = ELU( sum_m p[h,n,m]*Wh[h,m,o] )
// block: 8 waves = 16 n x 128 o; m-loop steps of 32, p-tile via LDS -> WMMA A
// ---------------------------------------------------------------------------
__global__ void k_agg1(const __bf16* __restrict__ WhT, const float* __restrict__ adj,
                       const float* __restrict__ f1, const float* __restrict__ f2,
                       const float* __restrict__ rmax, const float* __restrict__ rsum,
                       __bf16* __restrict__ hcat) {
    const int og = blockIdx.x, n0 = blockIdx.y * 16, h = blockIdx.z;
    const int tid = threadIdx.x, lane = tid & 31, w = tid >> 5;
    const int col0 = h * OO + og * 128 + w * 16;
    const float* f1h = f1 + h * NN;
    const float* f2h = f2 + h * NN;
    const float* rmh = rmax + h * NN;
    const float* rsh = rsum + h * NN;

    __shared__ __attribute__((aligned(16))) __bf16 pt[16][32];

    // each thread fills two p-tile entries: rows rA and rA+8, column lane
    const int rA = tid >> 5;
    const int nA = n0 + rA, nB = nA + 8;
    const float fA = f1h[nA], mA = rmh[nA], iA = 1.f / rsh[nA];
    const float fB = f1h[nB], mB = rmh[nB], iB = 1.f / rsh[nB];

    const int kbA = (lane >> 4) * 8;
    const int kbB = (lane >> 4) * 8;
    const __bf16* brow = WhT + (size_t)(col0 + (lane & 15)) * NN;
    v8f acc = {};

    for (int m0 = 0; m0 < NN; m0 += 32) {
        int m = m0 + lane;
        float f2v = f2h[m];
        float aA = adj[(size_t)nA * NN + m];
        float aB = adj[(size_t)nB * NN + m];
        pt[rA][lane]     = (__bf16)((aA > 0.f) ? __expf(lrelu(fA + f2v) - mA) * iA : 0.f);
        pt[rA + 8][lane] = (__bf16)((aB > 0.f) ? __expf(lrelu(fB + f2v) - mB) * iB : 0.f);
        __syncthreads();
        __builtin_prefetch(brow + m0 + 64, 0, 1);
        int am = lane & 15;
        v16bf af = cat8(*(const v8bf*)&pt[am][kbA], *(const v8bf*)&pt[am][16 + kbA]);
        v16bf bf = cat8(*(const v8bf*)(brow + m0 + kbB),
                        *(const v8bf*)(brow + m0 + 16 + kbB));
        acc = wmma_bf16(af, bf, acc);
        __syncthreads();
    }
    int cl = lane & 15, rh = lane >> 4;
    for (int r = 0; r < 8; ++r) {
        int n = n0 + rh * 8 + r;
        float v = acc[r];
        v = v > 0.f ? v : (__expf(v) - 1.f);            // ELU
        hcat[(size_t)n * HO + col0 + cl] = (__bf16)v;
    }
}

// ---------------------------------------------------------------------------
// GEMM2: Wh2T[c][n] = sum_k hcat[n][k] * WoutT[c][k]
// block = 4 waves covering 16 n x 64 c
// ---------------------------------------------------------------------------
__global__ void k_gemm2(const __bf16* __restrict__ A,   // [NN][HO]
                        const __bf16* __restrict__ BT,  // [CC][HO]
                        __bf16* __restrict__ CT) {      // [CC][NN]
    int lane = threadIdx.x & 31, w = threadIdx.x >> 5;
    int c0 = w * 16;
    int n0 = blockIdx.x * 16;
    int kbA = (lane >> 4) * 8, kbB = (lane >> 4) * 8;
    const __bf16* arow = A  + (size_t)(n0 + (lane & 15)) * HO;
    const __bf16* brow = BT + (size_t)(c0 + (lane & 15)) * HO;
    v8f acc = {};
    for (int k0 = 0; k0 < HO; k0 += 32) {
        __builtin_prefetch(arow + k0 + 128, 0, 1);
        v16bf af = cat8(*(const v8bf*)(arow + k0 + kbA),
                        *(const v8bf*)(arow + k0 + 16 + kbA));
        v16bf bf = cat8(*(const v8bf*)(brow + k0 + kbB),
                        *(const v8bf*)(brow + k0 + 16 + kbB));
        acc = wmma_bf16(af, bf, acc);
    }
    int cl = lane & 15, rh = lane >> 4;
    for (int r = 0; r < 8; ++r) {
        int n = n0 + rh * 8 + r;
        CT[(size_t)(c0 + cl) * NN + n] = (__bf16)acc[r];
    }
}

// ---------------------------------------------------------------------------
// Fused aggregation layer 2 + tanh, writes the (code, code, code) tuple
// block: 4 waves = 16 n x 64 c
// ---------------------------------------------------------------------------
__global__ void k_agg2(const __bf16* __restrict__ Wh2T, const float* __restrict__ adj,
                       const float* __restrict__ f1, const float* __restrict__ f2,
                       const float* __restrict__ rmax, const float* __restrict__ rsum,
                       float* __restrict__ out) {
    const int n0 = blockIdx.x * 16;
    const int tid = threadIdx.x, lane = tid & 31, w = tid >> 5;
    const int c0 = w * 16;

    __shared__ __attribute__((aligned(16))) __bf16 pt[16][32];

    // 128 threads fill 512 entries: rows r0, r0+4, r0+8, r0+12
    const int r0 = tid >> 5;
    float fv[4], mv[4], iv[4];
    int nrow[4];
    for (int j = 0; j < 4; ++j) {
        int n = n0 + r0 + 4 * j;
        nrow[j] = n;
        fv[j] = f1[n]; mv[j] = rmax[n]; iv[j] = 1.f / rsum[n];
    }

    const int kbA = (lane >> 4) * 8;
    const int kbB = (lane >> 4) * 8;
    const __bf16* brow = Wh2T + (size_t)(c0 + (lane & 15)) * NN;
    v8f acc = {};

    for (int m0 = 0; m0 < NN; m0 += 32) {
        int m = m0 + lane;
        float f2v = f2[m];
        for (int j = 0; j < 4; ++j) {
            float av = adj[(size_t)nrow[j] * NN + m];
            pt[r0 + 4 * j][lane] =
                (__bf16)((av > 0.f) ? __expf(lrelu(fv[j] + f2v) - mv[j]) * iv[j] : 0.f);
        }
        __syncthreads();
        int am = lane & 15;
        v16bf af = cat8(*(const v8bf*)&pt[am][kbA], *(const v8bf*)&pt[am][16 + kbA]);
        v16bf bf = cat8(*(const v8bf*)(brow + m0 + kbB),
                        *(const v8bf*)(brow + m0 + 16 + kbB));
        acc = wmma_bf16(af, bf, acc);
        __syncthreads();
    }
    int cl = lane & 15, rh = lane >> 4;
    for (int r = 0; r < 8; ++r) {
        int n = n0 + rh * 8 + r;
        float t = tanhf(acc[r]);
        size_t o = (size_t)n * CC + c0 + cl;
        out[o] = t;
        out[o + (size_t)NN * CC] = t;
        out[o + 2 * (size_t)NN * CC] = t;
    }
}

// ---------------------------------------------------------------------------
extern "C" void kernel_launch(void* const* d_in, const int* in_sizes, int n_in,
                              void* d_out, int out_size, void* d_ws, size_t ws_size,
                              hipStream_t stream) {
    const float* x        = (const float*)d_in[0];
    const float* adj      = (const float*)d_in[1];
    const float* W_heads  = (const float*)d_in[2];
    const float* a1_heads = (const float*)d_in[3];
    const float* a2_heads = (const float*)d_in[4];
    const float* W_out    = (const float*)d_in[5];
    const float* a1_out   = (const float*)d_in[6];
    const float* a2_out   = (const float*)d_in[7];
    float* out = (float*)d_out;

    char* ws = (char*)d_ws;
    auto carve = [&](size_t bytes) -> char* {
        char* p = ws;
        ws += (bytes + 255) & ~(size_t)255;
        return p;
    };
    __bf16* xb    = (__bf16*)carve((size_t)NN * FPAD * 2);   // 11.5 MB
    __bf16* WcatT = (__bf16*)carve((size_t)HO * FPAD * 2);   // 23.1 MB
    __bf16* WhT   = (__bf16*)carve((size_t)HO * NN * 2);     // 67 MB
    __bf16* hcat  = (__bf16*)carve((size_t)NN * HO * 2);     // 67 MB
    __bf16* WoutT = (__bf16*)carve((size_t)CC * HO * 2);     // 1 MB
    __bf16* Wh2T  = (__bf16*)carve((size_t)CC * NN * 2);     // 0.5 MB
    float*  f1    = (float*)carve((size_t)HH * NN * 4);
    float*  f2    = (float*)carve((size_t)HH * NN * 4);
    float*  rmx   = (float*)carve((size_t)HH * NN * 4);
    float*  rsm   = (float*)carve((size_t)HH * NN * 4);
    float*  f1o   = (float*)carve((size_t)NN * 4);
    float*  f2o   = (float*)carve((size_t)NN * 4);
    float*  rmxo  = (float*)carve((size_t)NN * 4);
    float*  rsmo  = (float*)carve((size_t)NN * 4);

    k_convert_x<<<(NN * FPAD + 255) / 256, 256, 0, stream>>>(x, xb);
    k_convert_W<<<(HO * FPAD + 255) / 256, 256, 0, stream>>>(W_heads, WcatT);
    k_convert_Wout<<<(CC * HO + 255) / 256, 256, 0, stream>>>(W_out, WoutT);

    k_gemm1<<<dim3(HO / 128, NN / 16), 256, 0, stream>>>(xb, WcatT, WhT);
    k_f1f2_h<<<dim3(NN / 256, HH), 256, 0, stream>>>(WhT, a1_heads, a2_heads, f1, f2);
    k_rowstats<<<NN, 256, 0, stream>>>(adj, f1, f2, rmx, rsm, HH);
    k_agg1<<<dim3(8, NN / 16, HH), 256, 0, stream>>>(WhT, adj, f1, f2, rmx, rsm, hcat);

    k_gemm2<<<NN / 16, 128, 0, stream>>>(hcat, WoutT, Wh2T);
    k_f1f2_out<<<NN / 256, 256, 0, stream>>>(Wh2T, a1_out, a2_out, f1o, f2o);
    k_rowstats<<<NN, 256, 0, stream>>>(adj, f1o, f2o, rmxo, rsmo, 1);
    k_agg2<<<NN / 16, 128, 0, stream>>>(Wh2T, adj, f1o, f2o, rmxo, rsmo, out);
}
